// RNNBlock_24266565222939
// MI455X (gfx1250) — matrix operand order repaired and checked by reading, not compiled, not run on hardware
//
#include <hip/hip_runtime.h>

// Fused reversed-SimpleRNN + Dense head for MI455X (gfx1250, wave32, WMMA).
//
// 256 workgroups x 128 batch rows; recurrence fully on-chip:
//   LDS wt[256][296] bf16 : [w_rec^T | w_in^T | b_rnn | pad]  (K = 288 used)
//   LDS hx[128][296] bf16 : [h       | x_t    | 1.0   | pad]
//   h_new = relu( [h, x_t, 1] @ [w_rec; w_in; b]^T ) via v_wmma_f32_16x16x32_bf16.
// This round: software-pipelined LDS fragment loads (A double-buffered,
// B prefetched one kc ahead) and register-prefetch of x_{t-1} across the GEMM
// so HBM latency is off the 79-step critical path; native bf16 casts.

#define TSTEPS 79
#define FDIM   16
#define CDIM   256
#define BM     128        // batch rows per workgroup
#define NKC    9          // K chunks of 32 (256 h + 16 x + 1 bias + pad)
#define STRIDE 296        // bf16 elements per LDS row (conflict-free b128)

typedef __attribute__((ext_vector_type(16))) __bf16       v16bf;
typedef __attribute__((ext_vector_type(8)))  float        v8f;
typedef __attribute__((ext_vector_type(4)))  float        f4v;
typedef __attribute__((ext_vector_type(4)))  unsigned int u4v;

union FragU { v16bf bf; u4v u[2]; };

__device__ __forceinline__ unsigned short f2bf(float f) {
  return __builtin_bit_cast(unsigned short, static_cast<__bf16>(f));
}

#define WMMA_BF16(A, B, C) \
  __builtin_amdgcn_wmma_f32_16x16x32_bf16(false, (A), false, (B), (short)0, (C), false, false)

// 128x256 GEMM over K=288, software pipelined:
//   A fragment double-buffered (prefetch mt+1 / (0,kc+1) before WMMAs on mt),
//   B fragments for kc+1 prefetched at top of kc loop.
__device__ __forceinline__ void gemm_tile(const unsigned short* wt,
                                          const unsigned short* hx,
                                          int lane, int n0, v8f acc[8][2]) {
  const int aRow  = lane & 15;              // A: lane -> row M
  const int aKoff = (lane >> 4) << 3;       // lanes 16-31 hold K+8
  const int bRow  = n0 + (lane & 15);       // B: lane -> column N
  const int bKoff = (lane >> 4) << 3;       // lanes 16-31 hold K+8

  const unsigned short* bp0 = wt + bRow * STRIDE + bKoff;
  const unsigned short* ap0 = hx + aRow * STRIDE + aKoff;

  FragU b0, b1, a;
  b0.u[0] = *(const u4v*)(bp0);
  b0.u[1] = *(const u4v*)(bp0 + 16);
  b1.u[0] = *(const u4v*)(bp0 + 16 * STRIDE);
  b1.u[1] = *(const u4v*)(bp0 + 16 * STRIDE + 16);
  a.u[0]  = *(const u4v*)(ap0);
  a.u[1]  = *(const u4v*)(ap0 + 16);

#pragma unroll
  for (int kc = 0; kc < NKC; ++kc) {
    FragU b0n, b1n;
    if (kc + 1 < NKC) {                     // prefetch B for next kc
      const unsigned short* bp = bp0 + (kc + 1) * 32;
      b0n.u[0] = *(const u4v*)(bp);
      b0n.u[1] = *(const u4v*)(bp + 16);
      b1n.u[0] = *(const u4v*)(bp + 16 * STRIDE);
      b1n.u[1] = *(const u4v*)(bp + 16 * STRIDE + 16);
    }
#pragma unroll
    for (int mt = 0; mt < 8; ++mt) {
      FragU an;
      if (mt < 7) {                         // prefetch A for next mt
        const unsigned short* ap = ap0 + (mt + 1) * 16 * STRIDE + kc * 32;
        an.u[0] = *(const u4v*)(ap);
        an.u[1] = *(const u4v*)(ap + 16);
      } else if (kc + 1 < NKC) {            // prefetch A(0, kc+1)
        const unsigned short* ap = ap0 + (kc + 1) * 32;
        an.u[0] = *(const u4v*)(ap);
        an.u[1] = *(const u4v*)(ap + 16);
      }
      acc[mt][0] = WMMA_BF16(a.bf, b0.bf, acc[mt][0]);
      acc[mt][1] = WMMA_BF16(a.bf, b1.bf, acc[mt][1]);
      if (mt < 7 || kc + 1 < NKC) a = an;
    }
    if (kc + 1 < NKC) { b0 = b0n; b1 = b1n; }
  }
}

__device__ __forceinline__ void load_x(const float* __restrict__ x, int rowBlk,
                                       int tid, int t, f4v& a, f4v& b) {
  const int r = tid >> 1, half = tid & 1;
  const float* xp = x + ((size_t)(rowBlk + r) * TSTEPS + t) * FDIM + half * 8;
  a = *(const f4v*)xp;
  b = *(const f4v*)(xp + 4);
}

__device__ __forceinline__ void store_x(unsigned short* hx, int tid, f4v a, f4v b) {
  const int r = tid >> 1, half = tid & 1;
  u4v s;
  s[0] = (unsigned)f2bf(a[0]) | ((unsigned)f2bf(a[1]) << 16);
  s[1] = (unsigned)f2bf(a[2]) | ((unsigned)f2bf(a[3]) << 16);
  s[2] = (unsigned)f2bf(b[0]) | ((unsigned)f2bf(b[1]) << 16);
  s[3] = (unsigned)f2bf(b[2]) | ((unsigned)f2bf(b[3]) << 16);
  *(u4v*)(hx + r * STRIDE + 256 + half * 8) = s;
}

extern "C" __global__ __launch_bounds__(256)
void rnn_fused_kernel(const float* __restrict__ x,     const float* __restrict__ w_in,
                      const float* __restrict__ w_rec, const float* __restrict__ b_rnn,
                      const float* __restrict__ w_d,   const float* __restrict__ b_d,
                      float* __restrict__ out) {
  extern __shared__ char smem[];
  unsigned short* wt = (unsigned short*)smem;                       // [256][STRIDE]
  unsigned short* hx = (unsigned short*)(smem + CDIM * STRIDE * 2); // [128][STRIDE]

  const int tid    = threadIdx.x;
  const int lane   = tid & 31;
  const int wave   = tid >> 5;
  const int n0     = wave * 32;             // this wave's N strip
  const int rowBlk = blockIdx.x * BM;

  // ---- zero all LDS (h state, pads) ----
  {
    unsigned int* p = (unsigned int*)smem;
    const int n32 = (CDIM + BM) * STRIDE / 2;
    for (int i = tid; i < n32; i += 256) p[i] = 0u;
  }
  __syncthreads();

  // ---- stage weights: wt[n][k] = w_rec[k][n]; x-proj + bias appended in K ----
  for (int i = tid; i < CDIM * CDIM; i += 256) {
    int k = i >> 8, n = i & 255;
    wt[n * STRIDE + k] = f2bf(w_rec[i]);              // coalesced global read
  }
  for (int i = tid; i < FDIM * CDIM; i += 256) {
    int f = i >> 8, n = i & 255;
    wt[n * STRIDE + 256 + f] = f2bf(w_in[i]);
  }
  if (tid < CDIM) wt[tid * STRIDE + 272] = f2bf(b_rnn[tid]);
  if (tid < BM)   hx[tid * STRIDE + 272] = 0x3F80;    // bias multiplier = 1.0
  // prologue: stage x_{T-1}
  {
    f4v xa, xb;
    load_x(x, rowBlk, tid, TSTEPS - 1, xa, xb);
    store_x(hx, tid, xa, xb);
  }
  __syncthreads();

  const int cCol    = n0 + (lane & 15);               // C/D layout: lane -> N
  const int cRowOff = (lane >> 4) << 3;               // lanes 16-31 -> M+8

  // ---- reversed-time recurrence, fully on-chip ----
  for (int t = TSTEPS - 1; t >= 0; --t) {
    // prefetch x_{t-1} into registers; the global load is in flight for the
    // whole GEMM (no conversion here, so no wait is forced at issue)
    f4v xa, xb;
    if (t > 0) load_x(x, rowBlk, tid, t - 1, xa, xb);

    v8f acc[8][2];
#pragma unroll
    for (int mt = 0; mt < 8; ++mt) {
      acc[mt][0] = (v8f){0.f, 0.f, 0.f, 0.f, 0.f, 0.f, 0.f, 0.f};
      acc[mt][1] = (v8f){0.f, 0.f, 0.f, 0.f, 0.f, 0.f, 0.f, 0.f};
    }
    gemm_tile(wt, hx, lane, n0, acc);
    __syncthreads();   // every wave's LDS reads done before hx is overwritten

    // h = relu(acc) -> bf16 (C/D-layout scatter); stage prefetched x_{t-1}
#pragma unroll
    for (int mt = 0; mt < 8; ++mt) {
#pragma unroll
      for (int nt = 0; nt < 2; ++nt) {
        v8f a = acc[mt][nt];
#pragma unroll
        for (int v = 0; v < 8; ++v) {
          float r = a[v] > 0.f ? a[v] : 0.f;
          hx[(mt * 16 + cRowOff + v) * STRIDE + cCol + nt * 16] = f2bf(r);
        }
      }
    }
    if (t > 0) store_x(hx, tid, xa, xb);
    __syncthreads();   // h_t (+ x_{t-1}) visible before next step's reads
  }

  // ---- dense head: rewrite wt with w_d^T + b_d, zero x-weight slots ----
  for (int i = tid; i < CDIM * CDIM; i += 256) {
    int k = i >> 8, n = i & 255;
    wt[n * STRIDE + k] = f2bf(w_d[i]);
  }
  for (int i = tid; i < FDIM * CDIM; i += 256) {
    int f = i >> 8, n = i & 255;
    wt[n * STRIDE + 256 + f] = 0;   // stale x_0 slots contribute 0
  }
  if (tid < CDIM) wt[tid * STRIDE + 272] = f2bf(b_d[tid]);
  __syncthreads();

  v8f acc[8][2];
#pragma unroll
  for (int mt = 0; mt < 8; ++mt) {
    acc[mt][0] = (v8f){0.f, 0.f, 0.f, 0.f, 0.f, 0.f, 0.f, 0.f};
    acc[mt][1] = (v8f){0.f, 0.f, 0.f, 0.f, 0.f, 0.f, 0.f, 0.f};
  }
  gemm_tile(wt, hx, lane, n0, acc);

  // out = relu(h_last @ w_d + b_d), fp32
#pragma unroll
  for (int mt = 0; mt < 8; ++mt) {
#pragma unroll
    for (int nt = 0; nt < 2; ++nt) {
      v8f a = acc[mt][nt];
#pragma unroll
      for (int v = 0; v < 8; ++v) {
        float r = a[v] > 0.f ? a[v] : 0.f;
        out[(size_t)(rowBlk + mt * 16 + cRowOff + v) * CDIM + cCol + nt * 16] = r;
      }
    }
  }
}

extern "C" void kernel_launch(void* const* d_in, const int* in_sizes, int n_in,
                              void* d_out, int out_size, void* d_ws, size_t ws_size,
                              hipStream_t stream) {
  const float* x     = (const float*)d_in[0];
  const float* w_in  = (const float*)d_in[1];
  const float* w_rec = (const float*)d_in[2];
  const float* b_rnn = (const float*)d_in[3];
  const float* w_d   = (const float*)d_in[4];
  const float* b_d   = (const float*)d_in[5];
  float* out = (float*)d_out;

  const int grid = 32768 / BM;                           // 256 workgroups
  const size_t shmem = (size_t)(CDIM + BM) * STRIDE * 2; // 227,328 B < 320 KB/WGP
  rnn_fused_kernel<<<grid, 256, shmem, stream>>>(x, w_in, w_rec, b_rnn, w_d, b_d, out);
}
